// Attention_36481452212797
// MI455X (gfx1250) — matrix operand order repaired and checked by reading, not compiled, not run on hardware
//
#include <hip/hip_runtime.h>
#include <math.h>

// ---------------- problem constants ----------------
constexpr int cB  = 8;
constexpr int cN  = 4096;
constexpr int cC  = 256;
constexpr int cH  = 8;
constexpr int cKC = 128;
constexpr int cC4 = 1024;
constexpr int cHD = 128;
constexpr int cBN = cB * cN;           // 32768
constexpr float cSCALE = 0.17677669529663687f;  // (C/H)^-0.5

// ---------------- WMMA plumbing (CDNA5 gfx1250, wave32) ----------------
typedef __bf16 bf16;
typedef __bf16 bf16x16 __attribute__((ext_vector_type(16)));
typedef float  f32x8   __attribute__((ext_vector_type(8)));

constexpr int TK  = 64;   // K elements staged in LDS per iteration (2 WMMA k-steps)
constexpr int LDT = 80;   // padded LDS row stride (elements): 64 + 16

__device__ __forceinline__ f32x8 wmma_bf16(bf16x16 a, bf16x16 b, f32x8 c) {
  // D(16x16,f32) = A(16x32,bf16) x B(32x16,bf16) + C
  return __builtin_amdgcn_wmma_f32_16x16x32_bf16(false, a, false, b, (short)0, c,
                                                 false, false);
}

union FragU  { bf16x16 v; uint4 q4[2]; bf16 e[16]; };
union Chunk8 { uint4 q;   bf16 e[8]; };

// A fragment (16x32 bf16). lane<16 -> M=lane, K={kb..kb+7, kb+16..kb+23}, kb=0;
// lane>=16 -> kb=8.
__device__ __forceinline__ bf16x16 load_fragA(const bf16* sA_strip, int lane) {
  int m  = lane & 15;
  int kb = (lane >> 4) << 3;
  const bf16* p = sA_strip + m * LDT + kb;
  FragU f;
  f.q4[0] = *(const uint4*)(p);        // K = kb .. kb+7
  f.q4[1] = *(const uint4*)(p + 16);   // K = kb+16 .. kb+23
  return f.v;
}

// B fragment (32x16 bf16). lane<16 -> N=lane, K={0..7,16..23}; lane>=16 -> K={8..15,24..31}.
__device__ __forceinline__ bf16x16 load_fragB(const bf16* sB, int n0, int lane) {
  int n  = n0 + (lane & 15);
  int kb = (lane >> 4) << 3;
  const bf16* p = sB + n * LDT + kb;
  FragU f;
  f.q4[0] = *(const uint4*)(p);
  f.q4[1] = *(const uint4*)(p + 16);
  return f.v;
}

// One LDS K-step: A frag + B frags, then WMMAs.
template<int NT>
__device__ __forceinline__ void mma_step(const bf16* sA, const bf16* sB, int w,
                                         int lane, f32x8* acc) {
#pragma unroll
  for (int kk = 0; kk < TK; kk += 32) {
    bf16x16 a = load_fragA(sA + w * 16 * LDT + kk, lane);
    bf16x16 bfr[NT];
#pragma unroll
    for (int ct = 0; ct < NT; ++ct) bfr[ct] = load_fragB(sB + kk, ct * 16, lane);
#pragma unroll
    for (int ct = 0; ct < NT; ++ct) acc[ct] = wmma_bf16(a, bfr[ct], acc[ct]);
  }
}

// ---- CDNA5 async global->LDS copy (no VGPR staging; tracked by ASYNCcnt) --------
__device__ __forceinline__ void async_copy_b128(unsigned int lds_off,
                                                const void* gsrc) {
  unsigned long long ga = (unsigned long long)(uintptr_t)gsrc;
  asm volatile("global_load_async_to_lds_b128 %0, %1, off"
               :: "v"(lds_off), "v"(ga)
               : "memory");
}

__device__ __forceinline__ void wait_async() {
  asm volatile("s_wait_asynccnt 0x0" ::: "memory");
}

// Fill LDS tile [ROWS][TK] from row-major global (ld = ldg) via async copies.
template<int ROWS>
__device__ __forceinline__ void fill_async(bf16* s, const bf16* g, int ldg,
                                           int tid, int nth) {
#pragma unroll
  for (int t = tid; t < ROWS * 8; t += nth) {
    int row = t >> 3, c = (t & 7) << 3;
    unsigned int dst = (unsigned int)(uintptr_t)(s + row * LDT + c);
    async_copy_b128(dst, g + (size_t)row * ldg + c);
  }
}

// Fill LDS tile [COLS][TK] from a global tile of TK rows x COLS cols, transposing
// through VGPRs (async copy cannot transpose).
template<int COLS>
__device__ __forceinline__ void fill_trans(bf16* s, const bf16* g, int ldg,
                                           int tid, int nth) {
  constexpr int CH = COLS / 8;
#pragma unroll
  for (int t = tid; t < TK * CH; t += nth) {
    int sr = t / CH, co = (t % CH) << 3;
    Chunk8 u;
    u.q = *(const uint4*)(g + (size_t)sr * ldg + co);
#pragma unroll
    for (int j = 0; j < 8; ++j) s[(co + j) * LDT + sr] = u.e[j];
  }
}

// ---------------- elementwise convert ----------------
__global__ void k_f32_to_bf16(const float* __restrict__ src, bf16* __restrict__ dst,
                              int n) {
  int i = blockIdx.x * blockDim.x + threadIdx.x;
  if (i < n) dst[i] = (bf16)src[i];
}

// ---------------- generic bf16-out GEMM: Out[M][N] = A[M][K] * Bt[N][K]^T --------
// tile 64x64, 128 threads (4 waves); async-copy double-buffered LDS.
__global__ __launch_bounds__(128) void k_gemm_bf16(const bf16* __restrict__ A, int lda,
                                                   const bf16* __restrict__ Bt, int ldb,
                                                   bf16* __restrict__ Out, int ldo,
                                                   int K) {
  __shared__ __align__(16) bf16 sA[2][64 * LDT];
  __shared__ __align__(16) bf16 sB[2][64 * LDT];
  int tid = threadIdx.x, lane = tid & 31, w = tid >> 5;
  int m0 = blockIdx.x * 64, n0 = blockIdx.y * 64;
  const bf16* Ag = A + (size_t)m0 * lda;
  const bf16* Bg = Bt + (size_t)n0 * ldb;
  fill_async<64>(sA[0], Ag, lda, tid, 128);
  fill_async<64>(sB[0], Bg, ldb, tid, 128);
  wait_async();
  __syncthreads();
  f32x8 acc[4] = {};
  int cur = 0;
  for (int k0 = 0; k0 < K; k0 += TK) {
    bool more = (k0 + TK < K);
    if (more) {
      fill_async<64>(sA[cur ^ 1], Ag + k0 + TK, lda, tid, 128);
      fill_async<64>(sB[cur ^ 1], Bg + k0 + TK, ldb, tid, 128);
    }
    mma_step<4>(sA[cur], sB[cur], w, lane, acc);
    if (more) wait_async();
    __syncthreads();
    cur ^= 1;
  }
  int n = lane & 15, rb = (lane >> 4) << 3;
  bf16* o = Out + (size_t)(m0 + w * 16) * ldo + n0;
#pragma unroll
  for (int ct = 0; ct < 4; ++ct)
#pragma unroll
    for (int e = 0; e < 8; ++e)
      o[(size_t)(rb + e) * ldo + ct * 16 + n] = (bf16)acc[ct][e];
}

// ---------------- bases init: adaptive-maxpool(32) over N + column L2 norm -------
__global__ __launch_bounds__(256) void k_init_bases(const bf16* __restrict__ Xs,
                                                    bf16* __restrict__ bases,    // [B][C4][KC]
                                                    bf16* __restrict__ basesT) { // [B][KC][C4]
  __shared__ float red[256];
  int blk = blockIdx.x;
  int b = blk >> 7, k = blk & (cKC - 1);
  int n0 = k * (cN / cKC);  // window of 32
  const bf16* xp = Xs + (size_t)b * cN * cC4 + (size_t)n0 * cC4;
  float mv[4];
#pragma unroll
  for (int i = 0; i < 4; ++i) {
    int c = threadIdx.x + i * 256;
    float m = -3.0e38f;
    for (int j = 0; j < 32; ++j) m = fmaxf(m, (float)xp[(size_t)j * cC4 + c]);
    mv[i] = m;
  }
  float sq = mv[0] * mv[0] + mv[1] * mv[1] + mv[2] * mv[2] + mv[3] * mv[3];
  red[threadIdx.x] = sq;
  __syncthreads();
  for (int s = 128; s > 0; s >>= 1) {
    if (threadIdx.x < s) red[threadIdx.x] += red[threadIdx.x + s];
    __syncthreads();
  }
  float inv = 1.0f / (1e-6f + sqrtf(red[0]));
#pragma unroll
  for (int i = 0; i < 4; ++i) {
    int c = threadIdx.x + i * 256;
    bf16 v = (bf16)(mv[i] * inv);
    basesT[(size_t)b * cKC * cC4 + (size_t)k * cC4 + c] = v;
    bases[(size_t)b * cC4 * cKC + (size_t)c * cKC + k]  = v;
  }
}

// ---------------- EM GEMM 1: scores = Xs · bases, row-softmax -> z ---------------
// per b: M=4096, N=128, K=1024. 4 waves, wave owns 16 rows x 128 cols;
// in-wave softmax via shfl over 16-lane halves; async double-buffered LDS.
__global__ __launch_bounds__(128) void k_scores_softmax(
    const bf16* __restrict__ Xs, const bf16* __restrict__ basesT,
    bf16* __restrict__ Z, float* __restrict__ znorm2) {
  __shared__ __align__(16) bf16 sA[2][64 * LDT];
  __shared__ __align__(16) bf16 sB[2][128 * LDT];
  int tid = threadIdx.x, lane = tid & 31, w = tid >> 5;
  int b = blockIdx.z, m0 = blockIdx.x * 64;
  const bf16* Ag = Xs + (size_t)b * cN * cC4 + (size_t)m0 * cC4;
  const bf16* Bg = basesT + (size_t)b * cKC * cC4;
  fill_async<64>(sA[0], Ag, cC4, tid, 128);
  fill_async<128>(sB[0], Bg, cC4, tid, 128);
  wait_async();
  __syncthreads();
  f32x8 acc[8] = {};
  int cur = 0;
  for (int k0 = 0; k0 < cC4; k0 += TK) {
    bool more = (k0 + TK < cC4);
    if (more) {
      fill_async<64>(sA[cur ^ 1], Ag + k0 + TK, cC4, tid, 128);
      fill_async<128>(sB[cur ^ 1], Bg + k0 + TK, cC4, tid, 128);
    }
    mma_step<8>(sA[cur], sB[cur], w, lane, acc);
    if (more) wait_async();
    __syncthreads();
    cur ^= 1;
  }
  int n = lane & 15, rb = (lane >> 4) << 3;
  float colsq[8];
#pragma unroll
  for (int ct = 0; ct < 8; ++ct) colsq[ct] = 0.0f;
#pragma unroll
  for (int e = 0; e < 8; ++e) {
    float m = -3.0e38f;
#pragma unroll
    for (int ct = 0; ct < 8; ++ct) m = fmaxf(m, acc[ct][e]);
#pragma unroll
    for (int off = 8; off >= 1; off >>= 1) m = fmaxf(m, __shfl_xor(m, off, 16));
    float ex[8], s = 0.0f;
#pragma unroll
    for (int ct = 0; ct < 8; ++ct) { ex[ct] = __expf(acc[ct][e] - m); s += ex[ct]; }
#pragma unroll
    for (int off = 8; off >= 1; off >>= 1) s += __shfl_xor(s, off, 16);
    float inv = 1.0f / s;
    int row = m0 + w * 16 + rb + e;
    bf16* zp = Z + (size_t)b * cN * cKC + (size_t)row * cKC;
#pragma unroll
    for (int ct = 0; ct < 8; ++ct) {
      float zv = ex[ct] * inv;
      zp[ct * 16 + n] = (bf16)zv;
      colsq[ct] += zv * zv;
    }
  }
#pragma unroll
  for (int ct = 0; ct < 8; ++ct)
    atomicAdd(&znorm2[b * cKC + ct * 16 + n], colsq[ct]);
}

// ---------------- EM GEMM 2: raw = Xs^T · z  (+ column sumsq) --------------------
// per b: M=C4, N=KC, K=4096. Both operands k-major -> VGPR transpose into LDS,
// double-buffered with fill-after-compute; L2 warm-up via global_prefetch.
__global__ __launch_bounds__(128) void k_bases_update(const bf16* __restrict__ Xs,
                                                      const bf16* __restrict__ Z,
                                                      float* __restrict__ raw,
                                                      float* __restrict__ rawnorm2) {
  __shared__ __align__(16) bf16 sA[2][64 * LDT];
  __shared__ __align__(16) bf16 sB[2][128 * LDT];
  int tid = threadIdx.x, lane = tid & 31, w = tid >> 5;
  int b = blockIdx.z, m0 = blockIdx.x * 64;  // m = channel
  const bf16* Ag = Xs + (size_t)b * cN * cC4 + m0;  // rows = token k, ld C4
  const bf16* Zg = Z + (size_t)b * cN * cKC;        // rows = token k, ld KC
  fill_trans<64>(sA[0], Ag, cC4, tid, 128);
  fill_trans<128>(sB[0], Zg, cKC, tid, 128);
  __syncthreads();
  f32x8 acc[8] = {};
  int cur = 0;
  for (int k0 = 0; k0 < cN; k0 += TK) {
    bool more = (k0 + TK < cN);
    if (k0 + 2 * TK < cN) {  // warm L2 two tiles ahead
      __builtin_prefetch(
          Ag + (size_t)(k0 + 2 * TK + (tid >> 3)) * cC4 + ((tid & 7) << 3), 0, 1);
      __builtin_prefetch(
          Zg + (size_t)(k0 + 2 * TK + (tid >> 4)) * cKC + ((tid & 15) << 3), 0, 1);
    }
    mma_step<8>(sA[cur], sB[cur], w, lane, acc);
    if (more) {
      fill_trans<64>(sA[cur ^ 1], Ag + (size_t)(k0 + TK) * cC4, cC4, tid, 128);
      fill_trans<128>(sB[cur ^ 1], Zg + (size_t)(k0 + TK) * cKC, cKC, tid, 128);
    }
    __syncthreads();
    cur ^= 1;
  }
  int n = lane & 15, rb = (lane >> 4) << 3;
  float* rp = raw + (size_t)b * cC4 * cKC + (size_t)(m0 + w * 16) * cKC;
#pragma unroll
  for (int ct = 0; ct < 8; ++ct) {
    float sq = 0.0f;
#pragma unroll
    for (int e = 0; e < 8; ++e) {
      float v = acc[ct][e];
      rp[(size_t)(rb + e) * cKC + ct * 16 + n] = v;
      sq += v * v;
    }
    atomicAdd(&rawnorm2[b * cKC + ct * 16 + n], sq);
  }
}

// ---------------- normalize: fold z column norm + bases column norm --------------
__global__ void k_normalize_bases(const float* __restrict__ raw,
                                  const float* __restrict__ znorm2,
                                  const float* __restrict__ rawnorm2,
                                  bf16* __restrict__ bases,
                                  bf16* __restrict__ basesT) {
  int idx = blockIdx.x * 256 + threadIdx.x;  // over B*C4*KC = 1048576
  int k = idx & (cKC - 1);
  int c = (idx >> 7) & (cC4 - 1);
  int b = idx >> 17;
  float s  = 1.0f / (1e-6f + sqrtf(znorm2[b * cKC + k]));
  float fn = s / (1e-6f + s * sqrtf(rawnorm2[b * cKC + k]));
  float v  = raw[idx] * fn;
  bases[idx] = (bf16)v;  // raw is [b][c][k] == bases layout
  basesT[(size_t)b * cKC * cC4 + (size_t)k * cC4 + c] = (bf16)v;
}

// ---------------- per-head attention over clusters + softmax ---------------------
// one block per (b,h): 128x128 = qh(128xKC) x kh(KCx128)^T, scale, row-softmax.
__global__ __launch_bounds__(256) void k_att(const bf16* __restrict__ qb,
                                             const bf16* __restrict__ kb,
                                             bf16* __restrict__ att) {
  __shared__ __align__(16) bf16 sA[128 * LDT];
  __shared__ __align__(16) bf16 sB[128 * LDT];
  int tid = threadIdx.x, lane = tid & 31, w = tid >> 5;  // 8 waves
  int bh = blockIdx.x;
  int b = bh >> 3, h = bh & 7;
  const bf16* Ab = qb + (size_t)b * cC4 * cKC + (size_t)h * cHD * cKC;
  const bf16* Bb = kb + (size_t)b * cC4 * cKC + (size_t)h * cHD * cKC;
  f32x8 acc[8] = {};
  for (int k0 = 0; k0 < cKC; k0 += TK) {
    fill_async<128>(sA, Ab + k0, cKC, tid, 256);
    fill_async<128>(sB, Bb + k0, cKC, tid, 256);
    wait_async();
    __syncthreads();
    mma_step<8>(sA, sB, w, lane, acc);
    __syncthreads();
  }
  int n = lane & 15, rb = (lane >> 4) << 3;
  bf16* op = att + (size_t)bh * cHD * cHD + (size_t)(w * 16) * cHD;
#pragma unroll
  for (int e = 0; e < 8; ++e) {
    float m = -3.0e38f;
#pragma unroll
    for (int ct = 0; ct < 8; ++ct) m = fmaxf(m, acc[ct][e] * cSCALE);
#pragma unroll
    for (int off = 8; off >= 1; off >>= 1) m = fmaxf(m, __shfl_xor(m, off, 16));
    float ex[8], s = 0.0f;
#pragma unroll
    for (int ct = 0; ct < 8; ++ct) {
      ex[ct] = __expf(acc[ct][e] * cSCALE - m);
      s += ex[ct];
    }
#pragma unroll
    for (int off = 8; off >= 1; off >>= 1) s += __shfl_xor(s, off, 16);
    float inv = 1.0f / s;
#pragma unroll
    for (int ct = 0; ct < 8; ++ct)
      op[(size_t)(rb + e) * cHD + ct * 16 + n] = (bf16)(ex[ct] * inv);
  }
}

// ---------------- o[b,n,h*HD+d] = sum_e att[b,h,d,e] * v[b,n,h*HD+e] -------------
__global__ __launch_bounds__(128) void k_apply_att(const bf16* __restrict__ V,
                                                   const bf16* __restrict__ att,
                                                   bf16* __restrict__ O) {
  __shared__ __align__(16) bf16 sA[2][64 * LDT];
  __shared__ __align__(16) bf16 sB[2][128 * LDT];
  int tid = threadIdx.x, lane = tid & 31, w = tid >> 5;
  int bz = blockIdx.z;
  int b = bz >> 3, h = bz & 7;
  int m0 = blockIdx.x * 64;
  const bf16* Ag = V + (size_t)b * cN * cC4 + (size_t)m0 * cC4 + h * cHD;  // [n][e]
  const bf16* Bg = att + (size_t)bz * cHD * cHD;                           // Bt [d][e]
  fill_async<64>(sA[0], Ag, cC4, tid, 128);
  fill_async<128>(sB[0], Bg, cHD, tid, 128);
  wait_async();
  __syncthreads();
  f32x8 acc[8] = {};
  int cur = 0;
  for (int k0 = 0; k0 < cHD; k0 += TK) {
    bool more = (k0 + TK < cHD);
    if (more) {
      fill_async<64>(sA[cur ^ 1], Ag + k0 + TK, cC4, tid, 128);
      fill_async<128>(sB[cur ^ 1], Bg + k0 + TK, cHD, tid, 128);
    }
    mma_step<8>(sA[cur], sB[cur], w, lane, acc);
    if (more) wait_async();
    __syncthreads();
    cur ^= 1;
  }
  int n = lane & 15, rb = (lane >> 4) << 3;
  bf16* o = O + (size_t)b * cN * cC4 + (size_t)(m0 + w * 16) * cC4 + h * cHD;
#pragma unroll
  for (int ct = 0; ct < 8; ++ct)
#pragma unroll
    for (int e = 0; e < 8; ++e)
      o[(size_t)(rb + e) * cC4 + ct * 16 + n] = (bf16)acc[ct][e];
}

// ---------------- final projection + bias + relu, f32 out ------------------------
__global__ __launch_bounds__(128) void k_out_proj(const bf16* __restrict__ A,
                                                  const bf16* __restrict__ Wpb,
                                                  const float* __restrict__ bias,
                                                  float* __restrict__ Out) {
  __shared__ __align__(16) bf16 sA[2][64 * LDT];
  __shared__ __align__(16) bf16 sB[2][64 * LDT];
  int tid = threadIdx.x, lane = tid & 31, w = tid >> 5;
  int m0 = blockIdx.x * 64, n0 = blockIdx.y * 64;
  const bf16* Ag = A + (size_t)m0 * cC4;
  const bf16* Bg = Wpb + (size_t)n0 * cC4;
  fill_async<64>(sA[0], Ag, cC4, tid, 128);
  fill_async<64>(sB[0], Bg, cC4, tid, 128);
  wait_async();
  __syncthreads();
  f32x8 acc[4] = {};
  int cur = 0;
  for (int k0 = 0; k0 < cC4; k0 += TK) {
    bool more = (k0 + TK < cC4);
    if (more) {
      fill_async<64>(sA[cur ^ 1], Ag + k0 + TK, cC4, tid, 128);
      fill_async<64>(sB[cur ^ 1], Bg + k0 + TK, cC4, tid, 128);
    }
    mma_step<4>(sA[cur], sB[cur], w, lane, acc);
    if (more) wait_async();
    __syncthreads();
    cur ^= 1;
  }
  int n = lane & 15, rb = (lane >> 4) << 3;
#pragma unroll
  for (int ct = 0; ct < 4; ++ct) {
    float bv = bias[n0 + ct * 16 + n];
#pragma unroll
    for (int e = 0; e < 8; ++e) {
      float v = acc[ct][e] + bv;
      Out[(size_t)(m0 + w * 16 + rb + e) * cC + n0 + ct * 16 + n] = fmaxf(v, 0.0f);
    }
  }
}

// ---------------- host orchestration ----------------
extern "C" void kernel_launch(void* const* d_in, const int* in_sizes, int n_in,
                              void* d_out, int out_size, void* d_ws, size_t ws_size,
                              hipStream_t stream) {
  (void)in_sizes; (void)n_in; (void)out_size;
  const float* x  = (const float*)d_in[0];
  const float* Wq = (const float*)d_in[1];
  const float* Wk = (const float*)d_in[2];
  const float* Wv = (const float*)d_in[3];
  const float* Wp = (const float*)d_in[4];
  const float* bp = (const float*)d_in[5];

  char* ws = (char*)d_ws;
  size_t off = 0;
  auto alloc = [&](size_t bytes) -> void* {
    void* p = ws + off;
    off = (off + bytes + 255) & ~(size_t)255;
    return p;
  };
  bf16* xb  = (bf16*)alloc((size_t)cBN * cC * 2);
  bf16* wqb = (bf16*)alloc((size_t)cC4 * cC * 2);
  bf16* wkb = (bf16*)alloc((size_t)cC4 * cC * 2);
  bf16* wvb = (bf16*)alloc((size_t)cC4 * cC * 2);
  bf16* wpb = (bf16*)alloc((size_t)cC * cC4 * 2);
  bf16* qbuf = (bf16*)alloc((size_t)cBN * cC4 * 2);
  bf16* kbuf = (bf16*)alloc((size_t)cBN * cC4 * 2);
  bf16* vbuf = (bf16*)alloc((size_t)cBN * cC4 * 2);
  bf16* z    = (bf16*)alloc((size_t)cBN * cKC * 2);
  float* raw = (float*)alloc((size_t)cB * cC4 * cKC * 4);
  float* znorm2   = (float*)alloc((size_t)cB * cKC * 4);
  float* rawnorm2 = (float*)alloc((size_t)cB * cKC * 4);
  bf16* basesq  = (bf16*)alloc((size_t)cB * cC4 * cKC * 2);
  bf16* basesqT = (bf16*)alloc((size_t)cB * cKC * cC4 * 2);
  bf16* basesk  = (bf16*)alloc((size_t)cB * cC4 * cKC * 2);
  bf16* baseskT = (bf16*)alloc((size_t)cB * cKC * cC4 * 2);
  bf16* att     = (bf16*)alloc((size_t)cB * cH * cHD * cHD * 2);
  if (off > ws_size) return;  // not enough scratch; deterministic no-op

  // 1) convert inputs to bf16
  k_f32_to_bf16<<<(cBN * cC + 255) / 256, 256, 0, stream>>>(x, xb, cBN * cC);
  k_f32_to_bf16<<<(cC4 * cC + 255) / 256, 256, 0, stream>>>(Wq, wqb, cC4 * cC);
  k_f32_to_bf16<<<(cC4 * cC + 255) / 256, 256, 0, stream>>>(Wk, wkb, cC4 * cC);
  k_f32_to_bf16<<<(cC4 * cC + 255) / 256, 256, 0, stream>>>(Wv, wvb, cC4 * cC);
  k_f32_to_bf16<<<(cC * cC4 + 255) / 256, 256, 0, stream>>>(Wp, wpb, cC * cC4);

  // 2) q/k/v projections: [BN,C] x [C4,C]^T
  dim3 gp(cBN / 64, cC4 / 64);
  k_gemm_bf16<<<gp, 128, 0, stream>>>(xb, cC, wqb, cC, qbuf, cC4, cC);
  k_gemm_bf16<<<gp, 128, 0, stream>>>(xb, cC, wkb, cC, kbuf, cC4, cC);
  k_gemm_bf16<<<gp, 128, 0, stream>>>(xb, cC, wvb, cC, vbuf, cC4, cC);

  // 3) bases init (maxpool + l2norm)
  k_init_bases<<<cB * cKC, 256, 0, stream>>>(qbuf, basesq, basesqT);
  k_init_bases<<<cB * cKC, 256, 0, stream>>>(kbuf, basesk, baseskT);

  // 4) EM stages
  for (int st = 0; st < 3; ++st) {
    for (int s = 0; s < 2; ++s) {
      const bf16* Xs = s ? kbuf : qbuf;
      bf16* bases  = s ? basesk : basesq;
      bf16* basesT = s ? baseskT : basesqT;
      hipMemsetAsync(znorm2, 0, (size_t)cB * cKC * 4, stream);
      hipMemsetAsync(rawnorm2, 0, (size_t)cB * cKC * 4, stream);
      k_scores_softmax<<<dim3(cN / 64, 1, cB), 128, 0, stream>>>(Xs, basesT, z,
                                                                 znorm2);
      k_bases_update<<<dim3(cC4 / 64, 1, cB), 128, 0, stream>>>(Xs, z, raw,
                                                                rawnorm2);
      k_normalize_bases<<<(cB * cC4 * cKC) / 256, 256, 0, stream>>>(raw, znorm2,
                                                                    rawnorm2, bases,
                                                                    basesT);
    }
  }

  // 5) per-head attention over clusters
  k_att<<<cB * cH, 256, 0, stream>>>(basesq, basesk, att);

  // 6) o = att @ v  (q buffer is dead now -> reuse as o)
  bf16* obuf = qbuf;
  k_apply_att<<<dim3(cN / 64, 1, cB * cH), 128, 0, stream>>>(vbuf, att, obuf);

  // 7) output projection + bias + relu (f32 out)
  k_out_proj<<<dim3(cBN / 64, cC / 64), 128, 0, stream>>>(obuf, wpb, bp,
                                                          (float*)d_out);
}
